// SiLUAttention_5248450036138
// MI455X (gfx1250) — matrix operand order repaired and checked by reading, not compiled, not run on hardware
//
#include <hip/hip_runtime.h>
#include <hip/hip_bf16.h>
#include <cstdint>

// ---------- types ----------
typedef __attribute__((ext_vector_type(16))) __bf16 v16bf;
typedef __attribute__((ext_vector_type(8)))  float  v8f;

struct FragBF { union { v16bf v; uint32_t u[8]; }; };

__device__ __forceinline__ unsigned short f2bf(float f) {
  uint32_t u = __float_as_uint(f);
  u = (u + 0x7FFFu + ((u >> 16) & 1u)) >> 16;
  return (unsigned short)u;
}
__device__ __forceinline__ float bf2f(unsigned short h) {
  return __uint_as_float(((uint32_t)h) << 16);
}
// fast silu: v_exp + v_rcp instead of IEEE division sequence
__device__ __forceinline__ float silu(float x) {
  return x * __builtin_amdgcn_rcpf(1.0f + __expf(-x));
}

// A-fragment K position for 16-bit 16x32 A (pairs): VGPR j, lane-half hh
__device__ __forceinline__ int a_kpos(int j, int hh) {
  return ((j & 3) * 2) + ((j >> 2) * 16) + hh * 8;
}
// B-fragment K position for 16-bit 32x16 B (pairs): VGPR j, lane-half hh
__device__ __forceinline__ int b_kpos(int j, int hh) {
  return ((j & 3) * 2) + ((j >> 2) * 16) + hh * 8;
}

// async memory->LDS copy, 8 bytes per lane, ASYNCcnt-tracked
__device__ __forceinline__ void async_b64(uint32_t ldsOff, uint32_t voff,
                                          const void* sbase) {
  asm volatile("global_load_async_to_lds_b64 %0, %1, %2"
               :: "v"(ldsOff), "v"(voff), "s"(sbase) : "memory");
}
__device__ __forceinline__ void wait_async0() {
  asm volatile("s_wait_asynccnt 0x0" ::: "memory");
}
__device__ __forceinline__ uint32_t lds_off32(const void* p) {
  return (uint32_t)(uintptr_t)p;   // low 32 bits of generic ptr = LDS byte offset
}

// ---------- constants ----------
#define DIM      1024
#define NHEADS   16
#define HD       64
#define SEQ      2048
#define MROWS    4096        // B*S
#define TSK      32          // K step per WMMA stage
#define LPAD     4           // LDS row pad (ushorts); keeps rows 8B-aligned

// ---------- kernel 0a: fp32 -> bf16 ----------
__global__ void cvt_bf16_kernel(const float* __restrict__ src,
                                unsigned short* __restrict__ dst, int n) {
  int i = blockIdx.x * 256 + threadIdx.x;
  if (i < n) dst[i] = f2bf(src[i]);
}

// ---------- kernel 0b: fp32 [K][N] -> bf16 [N][K] ----------
__global__ void cvt_t_bf16_kernel(const float* __restrict__ src,
                                  unsigned short* __restrict__ dst,
                                  int rows, int cols) {
  int i = blockIdx.x * 256 + threadIdx.x;
  if (i < rows * cols) {
    int r = i / cols, c = i % cols;
    dst[(size_t)c * rows + r] = f2bf(src[i]);
  }
}

// ---------- kernel 1: silu(X @ Wqkuv), scatter to q/k/v/gate [B,H,S,hd] bf16 ----------
__global__ __launch_bounds__(256)
void qkuv_gemm_kernel(const unsigned short* __restrict__ Xbf,  // [4096][1024]
                      const unsigned short* __restrict__ WT,   // [4096][1024] n-major
                      unsigned short* __restrict__ qOut,
                      unsigned short* __restrict__ kOut,
                      unsigned short* __restrict__ vOut,
                      unsigned short* __restrict__ gOut) {
  __shared__ unsigned short ldsX[128][TSK + LPAD];  // row stride 72B (8B aligned)
  __shared__ unsigned short ldsW[128][TSK + LPAD];

  const int tid  = threadIdx.x;
  const int lane = tid & 31;
  const int wid  = tid >> 5;        // 0..7
  const int hh   = lane >> 4;
  const int l16  = lane & 15;

  const int mBlock = blockIdx.y * 128;
  const int nBlock = blockIdx.x * 128;
  const int waveM  = (wid >> 2) * 64;   // 0,64
  const int waveN  = (wid & 3) * 32;    // 0..96

  const uint32_t ldsXb = lds_off32(&ldsX[0][0]);
  const uint32_t ldsWb = lds_off32(&ldsW[0][0]);
  const uint32_t rowB  = (TSK + LPAD) * 2;   // 72

  v8f acc[4][2];
  #pragma unroll
  for (int mt = 0; mt < 4; ++mt)
    #pragma unroll
    for (int nt = 0; nt < 2; ++nt) acc[mt][nt] = (v8f){};

  for (int k0 = 0; k0 < DIM; k0 += TSK) {
    // async fill: 128 rows x 64B per tile = 1024 b64 chunks; 256 thr x 4 each
    const void* srcX = (const void*)(Xbf + (size_t)mBlock * DIM + k0);
    const void* srcW = (const void*)(WT  + (size_t)nBlock * DIM + k0);
    #pragma unroll
    for (int i = 0; i < 4; ++i) {
      int idx = tid + i * 256;           // 0..1023
      int row = idx >> 3;
      int c8  = idx & 7;
      uint32_t voff = (uint32_t)row * (DIM * 2) + c8 * 8;
      uint32_t loff = (uint32_t)row * rowB + c8 * 8;
      async_b64(ldsXb + loff, voff, srcX);
      async_b64(ldsWb + loff, voff, srcW);
    }
    wait_async0();
    __syncthreads();

    FragBF a[4], b[2];
    #pragma unroll
    for (int mt = 0; mt < 4; ++mt) {
      int row = waveM + mt * 16 + l16;
      #pragma unroll
      for (int j = 0; j < 8; ++j)
        a[mt].u[j] = *(const uint32_t*)&ldsX[row][a_kpos(j, hh)];
    }
    #pragma unroll
    for (int nt = 0; nt < 2; ++nt) {
      int row = waveN + nt * 16 + l16;
      #pragma unroll
      for (int j = 0; j < 8; ++j)
        b[nt].u[j] = *(const uint32_t*)&ldsW[row][b_kpos(j, hh)];
    }
    #pragma unroll
    for (int mt = 0; mt < 4; ++mt)
      #pragma unroll
      for (int nt = 0; nt < 2; ++nt)
        acc[mt][nt] = __builtin_amdgcn_wmma_f32_16x16x32_bf16(
            false, a[mt].v, false, b[nt].v, (short)0, acc[mt][nt], false, false);
    __syncthreads();
  }

  // epilogue: silu + scatter into [B,H,S,hd]
  const int bIdx = mBlock >> 11;       // batch (uniform per block)
  unsigned short* outs[4] = {qOut, kOut, vOut, gOut};
  #pragma unroll
  for (int mt = 0; mt < 4; ++mt) {
    #pragma unroll
    for (int nt = 0; nt < 2; ++nt) {
      int nG    = nBlock + waveN + nt * 16;    // global col base (tile-uniform split)
      int which = nG >> 10;
      int cc    = nG & 1023;
      int h     = cc >> 6;
      int hd    = (cc & 63) + l16;
      unsigned short* dst =
          outs[which] + ((size_t)(bIdx * NHEADS + h) * SEQ) * HD + hd;
      #pragma unroll
      for (int r = 0; r < 8; ++r) {
        int m = mBlock + waveM + mt * 16 + r + hh * 8;
        int s = m & (SEQ - 1);
        dst[(size_t)s * HD] = f2bf(silu(acc[mt][nt][r]));
      }
    }
  }
}

// ---------- kernel 2: silu-attention + LayerNorm + gate -> Gated [4096][1024] bf16 ----------
__global__ __launch_bounds__(128)
void attn_kernel(const unsigned short* __restrict__ Q,
                 const unsigned short* __restrict__ Km,
                 const unsigned short* __restrict__ V,
                 const unsigned short* __restrict__ Gate,
                 const float* __restrict__ gamma,
                 const float* __restrict__ beta,
                 unsigned short* __restrict__ Gated) {
  __shared__ unsigned short ldsK[64][HD + LPAD];     // [key][hd], 136B rows
  __shared__ unsigned short ldsVT[HD][64 + LPAD];    // [hd][key]
  __shared__ unsigned short ldsS[4][16][64 + LPAD];  // per-wave scores [q][key]

  const int tid  = threadIdx.x;
  const int lane = tid & 31;
  const int wid  = tid >> 5;       // 0..3
  const int hh   = lane >> 4;
  const int l16  = lane & 15;

  const int bh    = blockIdx.y;            // 0..31
  const int qBase = blockIdx.x * 64;
  const size_t bhOff = (size_t)bh * SEQ * HD;

  const uint32_t ldsKb = lds_off32(&ldsK[0][0]);
  const uint32_t rowKB = (HD + LPAD) * 2;  // 136

  // Q fragments (rows fixed for the whole kernel)
  FragBF qf[2];
  {
    const unsigned short* qrow =
        Q + bhOff + (size_t)(qBase + wid * 16 + l16) * HD;
    #pragma unroll
    for (int s = 0; s < 2; ++s)
      #pragma unroll
      for (int j = 0; j < 8; ++j)
        qf[s].u[j] = *(const uint32_t*)&qrow[s * 32 + a_kpos(j, hh)];
  }

  v8f attAcc[4];
  #pragma unroll
  for (int nt = 0; nt < 4; ++nt) attAcc[nt] = (v8f){};

  const float scale = 0.125f;   // 1/sqrt(64)

  for (int c = 0; c < SEQ / 64; ++c) {
    const int keyBase = c * 64;
    // K chunk: async straight copy (64 rows x 128B = 1024 b64; 128 thr x 8)
    const void* srcK = (const void*)(Km + bhOff + (size_t)keyBase * HD);
    #pragma unroll
    for (int i = 0; i < 8; ++i) {
      int idx = tid + i * 128;
      int row = idx >> 4;
      int c8  = idx & 15;
      async_b64(ldsKb + (uint32_t)row * rowKB + c8 * 8,
                (uint32_t)row * (HD * 2) + c8 * 8, srcK);
    }
    // V chunk: regular load + transpose into LDS
    const uint32_t* srcV = (const uint32_t*)(V + bhOff + (size_t)keyBase * HD);
    #pragma unroll
    for (int i = 0; i < 16; ++i) {
      int idx = tid + i * 128;            // uint idx over [64][32]
      int row = idx >> 5;
      int c2  = idx & 31;
      uint32_t d = srcV[idx];
      ldsVT[c2 * 2][row]     = (unsigned short)(d & 0xFFFFu);
      ldsVT[c2 * 2 + 1][row] = (unsigned short)(d >> 16);
    }
    wait_async0();
    __syncthreads();

    // scores = silu(scale * Q K^T), written to per-wave LDS tile as bf16
    #pragma unroll
    for (int nt = 0; nt < 4; ++nt) {
      v8f sc = (v8f){};
      FragBF kb[2];
      int krow = nt * 16 + l16;
      #pragma unroll
      for (int s = 0; s < 2; ++s)
        #pragma unroll
        for (int j = 0; j < 8; ++j)
          kb[s].u[j] = *(const uint32_t*)&ldsK[krow][s * 32 + b_kpos(j, hh)];
      sc = __builtin_amdgcn_wmma_f32_16x16x32_bf16(
          false, qf[0].v, false, kb[0].v, (short)0, sc, false, false);
      sc = __builtin_amdgcn_wmma_f32_16x16x32_bf16(
          false, qf[1].v, false, kb[1].v, (short)0, sc, false, false);
      #pragma unroll
      for (int r = 0; r < 8; ++r)
        ldsS[wid][r + hh * 8][nt * 16 + l16] = f2bf(silu(sc[r] * scale));
    }

    // attended += weights @ V
    #pragma unroll
    for (int s = 0; s < 2; ++s) {
      FragBF af;
      #pragma unroll
      for (int j = 0; j < 8; ++j)
        af.u[j] = *(const uint32_t*)&ldsS[wid][l16][s * 32 + a_kpos(j, hh)];
      #pragma unroll
      for (int nt = 0; nt < 4; ++nt) {
        FragBF vb;
        int vrow = nt * 16 + l16;
        #pragma unroll
        for (int j = 0; j < 8; ++j)
          vb.u[j] = *(const uint32_t*)&ldsVT[vrow][s * 32 + b_kpos(j, hh)];
        attAcc[nt] = __builtin_amdgcn_wmma_f32_16x16x32_bf16(
            false, af.v, false, vb.v, (short)0, attAcc[nt], false, false);
      }
    }
    __syncthreads();
  }

  // LayerNorm over hd (spread across 16 lanes x 4 accumulators) + gate
  float g[4], bta[4];
  #pragma unroll
  for (int nt = 0; nt < 4; ++nt) {
    g[nt]   = gamma[nt * 16 + l16];
    bta[nt] = beta[nt * 16 + l16];
  }
  const int h = bh & (NHEADS - 1);
  const int b = bh >> 4;
  #pragma unroll
  for (int r = 0; r < 8; ++r) {
    float s1 = 0.f, s2 = 0.f;
    #pragma unroll
    for (int nt = 0; nt < 4; ++nt) {
      float x = attAcc[nt][r];
      s1 += x; s2 += x * x;
    }
    #pragma unroll
    for (int m = 1; m < 16; m <<= 1) {
      s1 += __shfl_xor(s1, m, 32);
      s2 += __shfl_xor(s2, m, 32);
    }
    const float mu  = s1 * (1.0f / HD);
    const float var = s2 * (1.0f / HD) - mu * mu;
    const float inv = __builtin_amdgcn_rsqf(var + 1e-5f);
    const int q = qBase + wid * 16 + r + hh * 8;
    const unsigned short* grow = Gate + bhOff + (size_t)q * HD;
    unsigned short* orow = Gated + (size_t)(b * SEQ + q) * DIM + h * HD;
    #pragma unroll
    for (int nt = 0; nt < 4; ++nt) {
      int hd = nt * 16 + l16;
      float x = (attAcc[nt][r] - mu) * inv * g[nt] + bta[nt];
      orow[hd] = f2bf(x * bf2f(grow[hd]));
    }
  }
}

// ---------- kernel 3: Gated @ Wout -> fp32 out ----------
__global__ __launch_bounds__(256)
void out_gemm_kernel(const unsigned short* __restrict__ Gbf,   // [4096][1024]
                     const unsigned short* __restrict__ WT,    // [1024][1024] n-major
                     float* __restrict__ Out) {                // [4096][1024]
  __shared__ unsigned short ldsX[128][TSK + LPAD];
  __shared__ unsigned short ldsW[128][TSK + LPAD];

  const int tid  = threadIdx.x;
  const int lane = tid & 31;
  const int wid  = tid >> 5;
  const int hh   = lane >> 4;
  const int l16  = lane & 15;

  const int mBlock = blockIdx.y * 128;
  const int nBlock = blockIdx.x * 128;
  const int waveM  = (wid >> 2) * 64;
  const int waveN  = (wid & 3) * 32;

  const uint32_t ldsXb = lds_off32(&ldsX[0][0]);
  const uint32_t ldsWb = lds_off32(&ldsW[0][0]);
  const uint32_t rowB  = (TSK + LPAD) * 2;   // 72

  v8f acc[4][2];
  #pragma unroll
  for (int mt = 0; mt < 4; ++mt)
    #pragma unroll
    for (int nt = 0; nt < 2; ++nt) acc[mt][nt] = (v8f){};

  for (int k0 = 0; k0 < DIM; k0 += TSK) {
    const void* srcX = (const void*)(Gbf + (size_t)mBlock * DIM + k0);
    const void* srcW = (const void*)(WT  + (size_t)nBlock * DIM + k0);
    #pragma unroll
    for (int i = 0; i < 4; ++i) {
      int idx = tid + i * 256;
      int row = idx >> 3;
      int c8  = idx & 7;
      uint32_t voff = (uint32_t)row * (DIM * 2) + c8 * 8;
      uint32_t loff = (uint32_t)row * rowB + c8 * 8;
      async_b64(ldsXb + loff, voff, srcX);
      async_b64(ldsWb + loff, voff, srcW);
    }
    wait_async0();
    __syncthreads();

    FragBF a[4], b[2];
    #pragma unroll
    for (int mt = 0; mt < 4; ++mt) {
      int row = waveM + mt * 16 + l16;
      #pragma unroll
      for (int j = 0; j < 8; ++j)
        a[mt].u[j] = *(const uint32_t*)&ldsX[row][a_kpos(j, hh)];
    }
    #pragma unroll
    for (int nt = 0; nt < 2; ++nt) {
      int row = waveN + nt * 16 + l16;
      #pragma unroll
      for (int j = 0; j < 8; ++j)
        b[nt].u[j] = *(const uint32_t*)&ldsW[row][b_kpos(j, hh)];
    }
    #pragma unroll
    for (int mt = 0; mt < 4; ++mt)
      #pragma unroll
      for (int nt = 0; nt < 2; ++nt)
        acc[mt][nt] = __builtin_amdgcn_wmma_f32_16x16x32_bf16(
            false, a[mt].v, false, b[nt].v, (short)0, acc[mt][nt], false, false);
    __syncthreads();
  }

  #pragma unroll
  for (int mt = 0; mt < 4; ++mt)
    #pragma unroll
    for (int nt = 0; nt < 2; ++nt) {
      int n = nBlock + waveN + nt * 16 + l16;
      #pragma unroll
      for (int r = 0; r < 8; ++r) {
        int m = mBlock + waveM + mt * 16 + r + hh * 8;
        Out[(size_t)m * DIM + n] = acc[mt][nt][r];
      }
    }
}

// ---------- launch ----------
extern "C" void kernel_launch(void* const* d_in, const int* in_sizes, int n_in,
                              void* d_out, int out_size, void* d_ws, size_t ws_size,
                              hipStream_t stream) {
  const float* hidden = (const float*)d_in[0];   // [2,2048,1024]
  const float* w_qkuv = (const float*)d_in[1];   // [1024,4096]
  const float* w_out  = (const float*)d_in[2];   // [1024,1024]
  const float* gamma  = (const float*)d_in[3];   // [64]
  const float* beta   = (const float*)d_in[4];   // [64]
  float* out = (float*)d_out;                    // [2,2048,1024]

  char* ws = (char*)d_ws;
  const size_t MB = 1u << 20;
  unsigned short* Xbf   = (unsigned short*)(ws + 0 * MB);   // 8 MiB
  unsigned short* WqT   = (unsigned short*)(ws + 8 * MB);   // 8 MiB
  unsigned short* WoT   = (unsigned short*)(ws + 16 * MB);  // 2 MiB
  unsigned short* Qb    = (unsigned short*)(ws + 18 * MB);  // 8 MiB
  unsigned short* Kb    = (unsigned short*)(ws + 26 * MB);  // 8 MiB
  unsigned short* Vb    = (unsigned short*)(ws + 34 * MB);  // 8 MiB
  unsigned short* Gb    = (unsigned short*)(ws + 42 * MB);  // 8 MiB
  unsigned short* Gated = (unsigned short*)(ws + 50 * MB);  // 8 MiB

  cvt_bf16_kernel<<<(MROWS * DIM) / 256, 256, 0, stream>>>(hidden, Xbf, MROWS * DIM);
  cvt_t_bf16_kernel<<<(DIM * 4 * DIM + 255) / 256, 256, 0, stream>>>(w_qkuv, WqT, DIM, 4 * DIM);
  cvt_t_bf16_kernel<<<(DIM * DIM + 255) / 256, 256, 0, stream>>>(w_out, WoT, DIM, DIM);

  dim3 g1(32, 32);   // N/128 x M/128
  qkuv_gemm_kernel<<<g1, 256, 0, stream>>>(Xbf, WqT, Qb, Kb, Vb, Gb);

  dim3 g2(SEQ / 64, 32);  // q-blocks x (b*h)
  attn_kernel<<<g2, 128, 0, stream>>>(Qb, Kb, Vb, Gb, gamma, beta, Gated);

  dim3 g3(8, 32);    // N/128 x M/128
  out_gemm_kernel<<<g3, 256, 0, stream>>>(Gated, WoT, out);
}
